// LocalLayer_26156350832858
// MI455X (gfx1250) — matrix operand order repaired and checked by reading, not compiled, not run on hardware
//
#include <hip/hip_runtime.h>
#include <hip/hip_bf16.h>

// LocalLayer: D=16 grid of units, K=8 patch, C=3, OUT=1000 codes, B=64, beta=10
#define DD    16
#define KK    8
#define CC    3
#define OUTN  1000
#define OUTP  1024          // padded to multiple of 32 (WMMA K / tiling)
#define BB    64
#define PP    192           // C*K*K
#define BETA  10.0f

typedef __attribute__((ext_vector_type(16))) __bf16 v16bf;
typedef __attribute__((ext_vector_type(8)))  __bf16 v8bf;
typedef __attribute__((ext_vector_type(8)))  float  v8f;

// ---- LDS layout (bytes) ----
// patches : [0,              24576)   bf16 [64][192]
// alds    : [24576,          155648)  bf16 [64][1024]  (logits, then z in place)
// wt      : [155648,         167936)  bf16 [192][32]   (W^T stage for GEMM2)
#define SM_PATCH 0
#define SM_A     24576
#define SM_WT    (24576 + 131072)
#define SM_TOTAL (24576 + 131072 + 12288)

// native conversions -> let the backend emit packed bf16 cvt hardware ops
__device__ __forceinline__ __bf16 f2bf(float f) { return (__bf16)f; }
__device__ __forceinline__ float  bf2f(__bf16 b) { return (float)b; }
__device__ __forceinline__ v8bf   cvt8(v8f f)  { return __builtin_convertvector(f, v8bf); }
__device__ __forceinline__ v8f    up8(v8bf b)  { return __builtin_convertvector(b, v8f); }

__global__ __launch_bounds__(512, 1)
void local_layer_kernel(const float* __restrict__ x,
                        const float* __restrict__ W,
                        float* __restrict__ out)
{
    extern __shared__ char smem[];
    __bf16* patches = (__bf16*)(smem + SM_PATCH);   // [64][192]
    __bf16* alds    = (__bf16*)(smem + SM_A);       // [64][1024]
    __bf16* wt      = (__bf16*)(smem + SM_WT);      // [192][32] transposed W chunk

    const int u    = blockIdx.x;          // unit id
    const int wu   = u >> 4;
    const int hu   = u & 15;
    const int tid  = threadIdx.x;
    const int lane = tid & 31;
    const int wid  = tid >> 5;            // 16 waves
    const int half = lane >> 4;           // lane group within wave
    const int l16  = lane & 15;

    const float* Wu = W + (size_t)u * OUTN * PP;

    // ---------------- phase 0: gather patches -> LDS bf16 ----------------
    // p = c*64 + kh*8 + kw  (channel-major flatten of the 8x8 patch)
    for (int e = tid; e < BB * PP; e += 512) {
        int b  = e / PP;
        int p  = e % PP;
        int c  = p >> 6;
        int kh = (p >> 3) & 7;
        int kw = p & 7;
        float v = x[(((size_t)b * CC + c) * 128 + (wu * 8 + kh)) * 128 + hu * 8 + kw];
        patches[e] = f2bf(v);
    }
    __syncthreads();

    // ---------------- phase 1: GEMM1  a[b,o] = patches . Wu[o,:] ----------------
    // N-tiles over o: 63 tiles of 16 (o 0..1007), M-tiles over batch: 4, K = 192 in 6x32
    for (int nt = wid; nt < 63; nt += 16) {
        const int  o  = nt * 16 + l16;
        const bool ov = (o < OUTN);
        v8f acc[4] = {};
        #pragma unroll
        for (int ks = 0; ks < 6; ++ks) {
            // B-fragment: lane n holds column o, K = ks*32 + half*16 + (0..15), contiguous
            v16bf bfrag;
            {
                v8f q0 = {}, q1 = {};
                if (ov) {
                    const v8f* w8 = (const v8f*)(Wu + (size_t)o * PP + ks * 32 + half * 16);
                    q0 = w8[0]; q1 = w8[1];
                }
                v8bf h0 = cvt8(q0);
                v8bf h1 = cvt8(q1);
                #pragma unroll
                for (int j = 0; j < 8; ++j) { bfrag[j] = h0[j]; bfrag[j + 8] = h1[j]; }
            }
            // A-fragments per M-tile: lane m holds row, K = ks*32 + half*16 + (0..15), contiguous
            #pragma unroll
            for (int m = 0; m < 4; ++m) {
                const __bf16* ar = patches + (m * 16 + l16) * PP + ks * 32;
                v8bf lo = *(const v8bf*)(ar + half * 16);
                v8bf hi = *(const v8bf*)(ar + half * 16 + 8);
                v16bf afrag;
                #pragma unroll
                for (int j = 0; j < 8; ++j) { afrag[j] = lo[j]; afrag[j + 8] = hi[j]; }
                acc[m] = __builtin_amdgcn_wmma_f32_16x16x32_bf16(
                    false, afrag, false, bfrag, (short)0, acc[m], false, false);
            }
        }
        // store logits tile -> LDS bf16; C/D layout: M = r + 8*half, N = l16
        #pragma unroll
        for (int m = 0; m < 4; ++m) {
            #pragma unroll
            for (int r = 0; r < 8; ++r) {
                int b = m * 16 + r + 8 * half;
                alds[b * OUTP + nt * 16 + l16] = f2bf(acc[m][r]);
            }
        }
    }
    __syncthreads();

    // ---------------- phase 2: softmax over o (beta=10), z bf16 in place ----------------
    for (int rr = 0; rr < 4; ++rr) {
        int b = wid * 4 + rr;
        __bf16* row = alds + b * OUTP;
        const int base = lane * 32;           // 32 contiguous elements per lane
        float v[32];
        #pragma unroll
        for (int ch = 0; ch < 4; ++ch) {
            v8bf d = *(const v8bf*)(row + base + ch * 8);
            v8f  f = up8(d);
            #pragma unroll
            for (int j = 0; j < 8; ++j) v[ch * 8 + j] = f[j];
        }
        float mx = -1e30f;
        #pragma unroll
        for (int i = 0; i < 32; ++i) {
            v[i] = (base + i < OUTN) ? BETA * v[i] : -1e30f;
            mx = fmaxf(mx, v[i]);
        }
        #pragma unroll
        for (int off = 16; off > 0; off >>= 1) mx = fmaxf(mx, __shfl_xor(mx, off, 32));
        float s = 0.0f;
        #pragma unroll
        for (int i = 0; i < 32; ++i) {
            float e = (v[i] > -1e29f) ? __expf(v[i] - mx) : 0.0f;
            v[i] = e;
            s += e;
        }
        #pragma unroll
        for (int off = 16; off > 0; off >>= 1) s += __shfl_xor(s, off, 32);
        float inv = 1.0f / s;
        #pragma unroll
        for (int ch = 0; ch < 4; ++ch) {
            v8f f;
            #pragma unroll
            for (int j = 0; j < 8; ++j) f[j] = v[ch * 8 + j] * inv;
            *(v8bf*)(row + base + ch * 8) = cvt8(f);
        }
    }
    __syncthreads();

    // ---------------- phase 3: GEMM2  pred[b,p] = z[b,:] . Wu[:,p] ----------------
    // 48 output tiles (4 M x 12 N), 3 per wave; K = o in 32 chunks of 32.
    v8f pacc[3] = {};
    const int t0 = wid * 3;
    for (int ks = 0; ks < 32; ++ks) {
        // stage W^T chunk: wt[p][oo] = Wu[ks*32+oo][p], coalesced read over p
        for (int e = tid; e < 32 * PP; e += 512) {
            int oo = e / PP;
            int p  = e % PP;
            int o  = ks * 32 + oo;
            float vv = (o < OUTN) ? Wu[(size_t)o * PP + p] : 0.0f;
            wt[p * 32 + oo] = f2bf(vv);
        }
        __syncthreads();
        #pragma unroll
        for (int i = 0; i < 3; ++i) {
            int t = t0 + i;
            int m = t / 12;
            int n = t % 12;
            // A-fragment from z
            const __bf16* zr = alds + (m * 16 + l16) * OUTP + ks * 32;
            v8bf lo = *(const v8bf*)(zr + half * 16);
            v8bf hi = *(const v8bf*)(zr + half * 16 + 8);
            v16bf afrag;
            #pragma unroll
            for (int j = 0; j < 8; ++j) { afrag[j] = lo[j]; afrag[j + 8] = hi[j]; }
            // B-fragment from staged W^T: lane n -> column p = n*16+l16, K contiguous
            const __bf16* br = wt + (n * 16 + l16) * 32 + half * 16;
            v8bf b0 = *(const v8bf*)(br);
            v8bf b1 = *(const v8bf*)(br + 8);
            v16bf bfrag;
            #pragma unroll
            for (int j = 0; j < 8; ++j) { bfrag[j] = b0[j]; bfrag[j + 8] = b1[j]; }
            pacc[i] = __builtin_amdgcn_wmma_f32_16x16x32_bf16(
                false, afrag, false, bfrag, (short)0, pacc[i], false, false);
        }
        __syncthreads();
    }

    // ---------------- phase 4: scatter pred tiles to output image ----------------
    #pragma unroll
    for (int i = 0; i < 3; ++i) {
        int t = t0 + i;
        int m = t / 12;
        int n = t % 12;
        int p  = n * 16 + l16;
        int c  = p >> 6;
        int kh = (p >> 3) & 7;
        int kw = p & 7;
        #pragma unroll
        for (int r = 0; r < 8; ++r) {
            int b = m * 16 + r + 8 * half;
            out[(((size_t)b * CC + c) * 128 + (wu * 8 + kh)) * 128 + hu * 8 + kw] = pacc[i][r];
        }
    }
}

extern "C" void kernel_launch(void* const* d_in, const int* in_sizes, int n_in,
                              void* d_out, int out_size, void* d_ws, size_t ws_size,
                              hipStream_t stream) {
    (void)in_sizes; (void)n_in; (void)d_ws; (void)ws_size; (void)out_size;
    const float* x = (const float*)d_in[0];   // (64, 3, 128, 128) fp32
    const float* W = (const float*)d_in[1];   // (16, 16, 1000, 192) fp32
    float* out = (float*)d_out;               // (64, 3, 128, 128) fp32

    // one workgroup per (w,h) unit; 512 threads = 16 wave32; 168 KB dynamic LDS
    local_layer_kernel<<<dim3(DD * DD), dim3(512), SM_TOTAL, stream>>>(x, W, out);
}
